// CoherenceGate_31653908972268
// MI455X (gfx1250) — matrix operand order repaired and checked by reading, not compiled, not run on hardware
//
#include <hip/hip_runtime.h>
#include <stdint.h>

typedef __attribute__((ext_vector_type(16))) __bf16 v16bf;
typedef __attribute__((ext_vector_type(8)))  float  v8f;
typedef __attribute__((ext_vector_type(4)))  int    v4i;

constexpr int Bdim = 8192;
constexpr int Kdim = 1024;
constexpr int BM   = 128;   // block tile rows (batch)
constexpr int BN   = 64;    // block tile cols (output k-index)
constexpr int KT   = 32;    // contraction slice (bf16 WMMA K)
constexpr int NKT  = Kdim / KT;
constexpr int LSTR = 40;    // padded LDS row stride in halves (80B, 16B-aligned)

#if __has_builtin(__builtin_amdgcn_global_load_async_to_lds_b128)
#define USE_ASYNC_LDS 1
#else
#define USE_ASYNC_LDS 0
#endif

// pack two fp32 into two bf16 (round-half-up) with a single v_perm_b32
__device__ __forceinline__ unsigned pack_bf16x2(float lo, float hi) {
    unsigned a = __float_as_uint(lo) + 0x8000u;
    unsigned b = __float_as_uint(hi) + 0x8000u;
    // result = { b[31:16], a[31:16] }
    return __builtin_amdgcn_perm(b, a, 0x07060302);
}

__device__ __forceinline__ void wait_async_copies() {
#if USE_ASYNC_LDS
#if __has_builtin(__builtin_amdgcn_s_wait_asynccnt)
    __builtin_amdgcn_s_wait_asynccnt(0);
#else
    asm volatile("s_wait_asynccnt 0x0" ::: "memory");
#endif
#endif
}

// ---------------- prepass: fp32 -> bf16 (bandwidth bound, hits L2) ----------
__global__ __launch_bounds__(256) void cvt_f32_bf16_kernel(
    const float* __restrict__ src, unsigned short* __restrict__ dst, int n4)
{
    int i = blockIdx.x * blockDim.x + threadIdx.x;
    if (i >= n4) return;
    float4 v = ((const float4*)src)[i];
    uint2 p;
    p.x = pack_bf16x2(v.x, v.y);
    p.y = pack_bf16x2(v.z, v.w);
    ((uint2*)dst)[i] = p;
}

// ---------------- fused dual-GEMM + sigmoid epilogue ------------------------
__global__ __launch_bounds__(256) void coherence_gate_kernel(
    const unsigned short* __restrict__ psiReB,   // bf16 psi_re  (B x K)
    const unsigned short* __restrict__ psiImB,   // bf16 psi_im  (B x K)
    const unsigned short* __restrict__ Wb,       // bf16 W       (K x K)
    const float* __restrict__ psi_re,            // fp32 originals for epilogue
    const float* __restrict__ psi_im,
    const float* __restrict__ bias,
    float* __restrict__ out)
{
    __shared__ __align__(16) unsigned short sAre[2][BM * LSTR];
    __shared__ __align__(16) unsigned short sAim[2][BM * LSTR];
    __shared__ __align__(16) unsigned short sWt [2][BN * LSTR];

    const int t     = threadIdx.x;
    const int lane  = t & 31;
    const int wid   = t >> 5;
    const int lr    = lane & 15;
    const int hi    = lane >> 4;
    const int waveM = wid & 3;     // 4 waves along M (4*32 = 128)
    const int waveN = wid >> 2;    // 2 waves along N (2*32 = 64)

    const int mBlock = blockIdx.y * BM;
    const int nBlock = blockIdx.x * BN;

    // copy assignment: 16B chunks (8 bf16) per lane
    // A tiles: 128 rows * 4 chunks = 512 -> 2 chunks/thread each
    const int ar0 = t >> 2,         ac0 = t & 3;
    const int ar1 = (t + 256) >> 2, ac1 = (t + 256) & 3;
    // W tile: 64 rows * 4 chunks = 256 -> 1 chunk/thread
    const int wr  = t >> 2,         wc  = t & 3;

    auto stage = [&](int buf, int jb) {
        const unsigned short* gRe0 = psiReB + (size_t)(mBlock + ar0) * Kdim + jb + ac0 * 8;
        const unsigned short* gRe1 = psiReB + (size_t)(mBlock + ar1) * Kdim + jb + ac1 * 8;
        const unsigned short* gIm0 = psiImB + (size_t)(mBlock + ar0) * Kdim + jb + ac0 * 8;
        const unsigned short* gIm1 = psiImB + (size_t)(mBlock + ar1) * Kdim + jb + ac1 * 8;
        const unsigned short* gW   = Wb     + (size_t)(nBlock + wr ) * Kdim + jb + wc  * 8;
        unsigned short* lRe0 = &sAre[buf][ar0 * LSTR + ac0 * 8];
        unsigned short* lRe1 = &sAre[buf][ar1 * LSTR + ac1 * 8];
        unsigned short* lIm0 = &sAim[buf][ar0 * LSTR + ac0 * 8];
        unsigned short* lIm1 = &sAim[buf][ar1 * LSTR + ac1 * 8];
        unsigned short* lW   = &sWt [buf][wr  * LSTR + wc  * 8];
#if USE_ASYNC_LDS
        // builtin signature (from clang diagnostics): (as(1) int4*, as(3) int4*, int, int)
        typedef __attribute__((address_space(1))) v4i* gap_t;
        typedef __attribute__((address_space(3))) v4i* lap_t;
        __builtin_amdgcn_global_load_async_to_lds_b128((gap_t)(v4i*)gRe0, (lap_t)(v4i*)lRe0, 0, 0);
        __builtin_amdgcn_global_load_async_to_lds_b128((gap_t)(v4i*)gRe1, (lap_t)(v4i*)lRe1, 0, 0);
        __builtin_amdgcn_global_load_async_to_lds_b128((gap_t)(v4i*)gIm0, (lap_t)(v4i*)lIm0, 0, 0);
        __builtin_amdgcn_global_load_async_to_lds_b128((gap_t)(v4i*)gIm1, (lap_t)(v4i*)lIm1, 0, 0);
        __builtin_amdgcn_global_load_async_to_lds_b128((gap_t)(v4i*)gW,   (lap_t)(v4i*)lW,   0, 0);
#else
        *(uint4*)lRe0 = *(const uint4*)gRe0;
        *(uint4*)lRe1 = *(const uint4*)gRe1;
        *(uint4*)lIm0 = *(const uint4*)gIm0;
        *(uint4*)lIm1 = *(const uint4*)gIm1;
        *(uint4*)lW   = *(const uint4*)gW;
#endif
    };

    // A fragment (16x32 bf16, ISA 7.12.2): lane(row=lr, hi):
    //   VGPR0..3 = K[hi*8 .. +7], VGPR4..7 = K[16+hi*8 .. +7]
    auto load_fragA = [&](const unsigned short* basePtr, int row) -> v16bf {
        union { uint4 q[2]; v16bf v; } f;
        const char* p = (const char*)(basePtr + row * LSTR);
        f.q[0] = *(const uint4*)(p + hi * 16);
        f.q[1] = *(const uint4*)(p + 32 + hi * 16);
        return f.v;
    };
    // B fragment (32x16 bf16): lane(col=lr, hi): VGPR0..3 = K[hi*8 .. +7], VGPR4..7 = K[16+hi*8 .. +7]
    auto load_fragB = [&](const unsigned short* basePtr, int col) -> v16bf {
        union { uint4 q[2]; v16bf v; } f;
        const char* p = (const char*)(basePtr + col * LSTR);
        f.q[0] = *(const uint4*)(p + hi * 16);
        f.q[1] = *(const uint4*)(p + 32 + hi * 16);
        return f.v;
    };

    v8f accRe[2][2] = {};
    v8f accIm[2][2] = {};

    stage(0, 0);
    wait_async_copies();
    __syncthreads();

    int cur = 0;
    for (int jt = 0; jt < NKT; ++jt) {
        const bool hasNext = (jt + 1) < NKT;
        if (hasNext) stage(cur ^ 1, (jt + 1) * KT);   // async, overlaps with math

        v16bf aR[2], aI[2], bF[2];
        #pragma unroll
        for (int sm = 0; sm < 2; ++sm) {
            const int row = waveM * 32 + sm * 16 + lr;
            aR[sm] = load_fragA(sAre[cur], row);
            aI[sm] = load_fragA(sAim[cur], row);
        }
        #pragma unroll
        for (int sn = 0; sn < 2; ++sn) {
            const int col = waveN * 32 + sn * 16 + lr;
            bF[sn] = load_fragB(sWt[cur], col);
        }

        #pragma unroll
        for (int sm = 0; sm < 2; ++sm) {
            #pragma unroll
            for (int sn = 0; sn < 2; ++sn) {
                accRe[sm][sn] = __builtin_amdgcn_wmma_f32_16x16x32_bf16(
                    false, aR[sm], false, bF[sn], (short)0, accRe[sm][sn], false, false);
                accIm[sm][sn] = __builtin_amdgcn_wmma_f32_16x16x32_bf16(
                    false, aI[sm], false, bF[sn], (short)0, accIm[sm][sn], false, false);
            }
        }

        wait_async_copies();
        __syncthreads();
        cur ^= 1;
    }

    // epilogue: gamma = sigmoid(psi_re*u_re + psi_im*u_im + bias)
    // C/D layout: VGPR r -> M = r + hi*8 ; lane lr -> N = lr
    #pragma unroll
    for (int sm = 0; sm < 2; ++sm) {
        #pragma unroll
        for (int sn = 0; sn < 2; ++sn) {
            const int n  = nBlock + waveN * 32 + sn * 16 + lr;
            const float bv = bias[n];
            #pragma unroll
            for (int r = 0; r < 8; ++r) {
                const int m = mBlock + waveM * 32 + sm * 16 + hi * 8 + r;
                const size_t off = (size_t)m * Kdim + n;
                const float x = psi_re[off] * accRe[sm][sn][r]
                              + psi_im[off] * accIm[sm][sn][r] + bv;
                out[off] = 1.0f / (1.0f + __expf(-x));
            }
        }
    }
}

extern "C" void kernel_launch(void* const* d_in, const int* in_sizes, int n_in,
                              void* d_out, int out_size, void* d_ws, size_t ws_size,
                              hipStream_t stream) {
    (void)in_sizes; (void)n_in; (void)out_size; (void)ws_size;
    const float* psi_re = (const float*)d_in[0];
    const float* psi_im = (const float*)d_in[1];
    const float* W      = (const float*)d_in[2];
    const float* bias   = (const float*)d_in[3];
    float* out          = (float*)d_out;

    // workspace layout: bf16 copies of psi_re, psi_im, W  (~34 MB, lives in L2)
    unsigned short* psiReB = (unsigned short*)d_ws;
    unsigned short* psiImB = psiReB + (size_t)Bdim * Kdim;
    unsigned short* Wb     = psiImB + (size_t)Bdim * Kdim;

    const int nPsi4 = Bdim * Kdim / 4;   // float4 count
    const int nW4   = Kdim * Kdim / 4;
    cvt_f32_bf16_kernel<<<nPsi4 / 256, 256, 0, stream>>>(psi_re, psiReB, nPsi4);
    cvt_f32_bf16_kernel<<<nPsi4 / 256, 256, 0, stream>>>(psi_im, psiImB, nPsi4);
    cvt_f32_bf16_kernel<<<nW4   / 256, 256, 0, stream>>>(W,      Wb,     nW4);

    dim3 grid(Kdim / BN, Bdim / BM);   // (16, 64)
    coherence_gate_kernel<<<grid, 256, 0, stream>>>(
        psiReB, psiImB, Wb, psi_re, psi_im, bias, out);
}
